// ESABotRGCN_4layers_32590211842601
// MI455X (gfx1250) — matrix operand order repaired and checked by reading, not compiled, not run on hardware
//
#include <hip/hip_runtime.h>
#include <hip/hip_bf16.h>

typedef __attribute__((ext_vector_type(16))) __bf16 v16bf;
typedef __attribute__((ext_vector_type(8)))  float  v8f;

__device__ __forceinline__ float lrelu(float v) { return v > 0.f ? v : 0.01f * v; }

enum { FUSE_NONE = 0, FUSE_BIAS = 1, FUSE_BIAS_LRELU = 2 };

// Load a 32-byte B fragment from a pointer with only 16-byte alignment.
__device__ __forceinline__ v16bf load_frag16(const __bf16* p)
{
    struct U { uint4 a, b; };
    U u;
    u.a = *(const uint4*)p;
    u.b = *(const uint4*)(p + 8);
    return __builtin_bit_cast(v16bf, u);
}

// Transpose + convert weight W[K][ncols] (f32 row-major) into
// Wt[ncolsPad][K] (bf16), zero-padded columns. Wt row n = column n of W.
__global__ void prep_weight(const float* __restrict__ W, int K, int ncols, int ncolsPad,
                            __bf16* __restrict__ Wt)
{
    int idx = blockIdx.x * blockDim.x + threadIdx.x;
    int total = ncolsPad * K;
    if (idx >= total) return;
    int n = idx / K;
    int k = idx - n * K;
    float v = (n < ncols) ? W[(long)k * ncols + n] : 0.f;
    Wt[(long)n * K + k] = (__bf16)v;
}

// C[row, col_off+col] = fuse(A[nrows,K] @ W[K,ncols] (+bias)), W given as
// pre-transposed bf16 Wt[NT*16][K]. One wave computes a 16 x (NT*16) tile:
// A fragment loaded once per K-step, NT back-to-back WMMAs.
// STAGE=true: whole B panel staged in (row-padded) LDS, fragments via ds_load.
// Requires K % 32 == 0. A row clamped once (padded rows/cols never stored).
template <int NT, bool STAGE>
__global__ __launch_bounds__(256)
void gemm_wmma(const float* __restrict__ A, int lda, int nrows, int K,
               const __bf16* __restrict__ Wt, int ncols,
               const float* __restrict__ bias,
               float* __restrict__ C, int ldc, int col_off, int fuse)
{
    extern __shared__ __align__(16) char smem_raw[];
    __bf16* sW = (__bf16*)smem_raw;
    const int SK = K + 8;                        // padded row stride (bank-conflict-free)

    if (STAGE) {
        const int chunksPerRow = K / 8;          // 16-byte chunks per row
        const int totalChunks  = NT * 16 * chunksPerRow;
        for (int i = threadIdx.x; i < totalChunks; i += blockDim.x) {
            int n = i / chunksPerRow;
            int c = (i - n * chunksPerRow) * 8;
            *(uint4*)(sW + (long)n * SK + c) = *(const uint4*)(Wt + (long)n * K + c);
        }
        __syncthreads();
    }

    const int lane = threadIdx.x & 31;
    const int wave = threadIdx.x >> 5;
    const int row0 = (blockIdx.x * (blockDim.x >> 5) + wave) * 16;
    if (row0 >= nrows) return;

    const int m  = lane & 15;   // A-row / B-col within a 16-tile
    const int hw = lane >> 4;   // half-wave select

    long arow = row0 + m;
    if (arow > nrows - 1) arow = nrows - 1;
    const float*  Arow  = A + arow * lda;
    const __bf16* Wlane = Wt + (long)m * K;          // global path
    const __bf16* Slane = sW + (long)m * SK;         // LDS path

    v8f acc[NT];
    #pragma unroll
    for (int t = 0; t < NT; ++t) { v8f z = {}; acc[t] = z; }

    for (int k0 = 0; k0 < K; k0 += 32) {
        // A 16x32 bf16 fragment: a[0..15] = K k0+hw*16+0..15
        const float4 f0 = *(const float4*)(Arow + k0 + hw * 16);
        const float4 f1 = *(const float4*)(Arow + k0 + hw * 16 + 4);
        const float4 f2 = *(const float4*)(Arow + k0 + hw * 16 + 8);
        const float4 f3 = *(const float4*)(Arow + k0 + hw * 16 + 12);
        v16bf a;
        a[0]  = (__bf16)f0.x; a[1]  = (__bf16)f0.y; a[2]  = (__bf16)f0.z; a[3]  = (__bf16)f0.w;
        a[4]  = (__bf16)f1.x; a[5]  = (__bf16)f1.y; a[6]  = (__bf16)f1.z; a[7]  = (__bf16)f1.w;
        a[8]  = (__bf16)f2.x; a[9]  = (__bf16)f2.y; a[10] = (__bf16)f2.z; a[11] = (__bf16)f2.w;
        a[12] = (__bf16)f3.x; a[13] = (__bf16)f3.y; a[14] = (__bf16)f3.z; a[15] = (__bf16)f3.w;

        #pragma unroll
        for (int t = 0; t < NT; ++t) {
            const v16bf b = STAGE
                ? load_frag16(Slane + (long)t * 16 * SK + k0 + hw * 16)
                : load_frag16(Wlane + (long)t * 16 * K  + k0 + hw * 16);
            acc[t] = __builtin_amdgcn_wmma_f32_16x16x32_bf16(
                false, a, false, b, (short)0, acc[t], false, false);
        }
    }

    #pragma unroll
    for (int t = 0; t < NT; ++t) {
        const int col = t * 16 + m;
        if (col >= ncols) continue;
        float bv = (fuse != FUSE_NONE && bias) ? bias[col] : 0.f;
        #pragma unroll
        for (int g = 0; g < 8; ++g) {
            const int row = row0 + hw * 8 + g;
            if (row < nrows) {
                float v = acc[t][g] + bv;
                if (fuse == FUSE_BIAS_LRELU) v = lrelu(v);
                C[(long)row * ldc + col_off + col] = v;
            }
        }
    }
}

// Finish encoder: bias+lrelu on des/tweet cols, small GEMMs into cols 64..127.
__global__ void encode_small(const float* __restrict__ num, const float* __restrict__ cat,
                             const float* __restrict__ nf,
                             const float* __restrict__ Wn, const float* __restrict__ bn,
                             const float* __restrict__ Wc, const float* __restrict__ bc,
                             const float* __restrict__ Ww, const float* __restrict__ bw,
                             const float* __restrict__ bdes, const float* __restrict__ btw,
                             float* __restrict__ x, int nrows)
{
    int i = blockIdx.x * blockDim.x + threadIdx.x;
    if (i >= nrows) return;
    float* row = x + (long)i * 128;
    #pragma unroll 4
    for (int c = 0; c < 28; ++c) row[c] = lrelu(row[c] + bdes[c]);
    #pragma unroll 4
    for (int c = 0; c < 36; ++c) row[28 + c] = lrelu(row[28 + c] + btw[c]);

    float nin[7];
    #pragma unroll
    for (int k = 0; k < 7; ++k) nin[k] = num[(long)i * 7 + k];
    for (int c = 0; c < 12; ++c) {
        float s = bn[c];
        #pragma unroll
        for (int k = 0; k < 7; ++k) s += nin[k] * Wn[k * 12 + c];
        row[64 + c] = lrelu(s);
    }
    float cin[11];
    #pragma unroll
    for (int k = 0; k < 11; ++k) cin[k] = cat[(long)i * 11 + k];
    for (int c = 0; c < 40; ++c) {
        float s = bc[c];
        #pragma unroll
        for (int k = 0; k < 11; ++k) s += cin[k] * Wc[k * 40 + c];
        row[76 + c] = lrelu(s);
    }
    float nv = nf[i];
    for (int c = 0; c < 12; ++c) row[116 + c] = lrelu(bw[c] + nv * Ww[c]);
}

__global__ void zero_f32(float* __restrict__ p, int n)
{
    int i = blockIdx.x * blockDim.x + threadIdx.x;
    if (i < n) p[i] = 0.f;
}

__global__ void degree_kernel(const int* __restrict__ dst, const int* __restrict__ et,
                              float* __restrict__ cnt, int E, int nrows)
{
    int e = blockIdx.x * blockDim.x + threadIdx.x;
    if (e >= E) return;
    atomicAdd(&cnt[(long)et[e] * nrows + dst[e]], 1.0f);
}

__global__ void inv_kernel(float* __restrict__ cnt, int n)
{
    int i = blockIdx.x * blockDim.x + threadIdx.x;
    if (i < n) cnt[i] = 1.0f / fmaxf(cnt[i], 1.0f);
}

// One wave per edge: out[dst] += (x @ W_r)[src] * inv[r][dst].
// Each lane owns 4 contiguous columns -> float4 load + 4 f32 atomics.
__global__ void scatter_kernel(const int* __restrict__ src, const int* __restrict__ dst,
                               const int* __restrict__ et,
                               const float* __restrict__ xr0, const float* __restrict__ xr1,
                               const float* __restrict__ inv,
                               float* __restrict__ out, int E, int nrows)
{
    int e = blockIdx.x * (blockDim.x >> 5) + (threadIdx.x >> 5);
    if (e >= E) return;
    int lane = threadIdx.x & 31;
    int s = src[e], d = dst[e], r = et[e];
    const float* mrow = (r ? xr1 : xr0) + (long)s * 128;
    float* orow = out + (long)d * 128;
    float sc = inv[(long)r * nrows + d];
    const float4 mv = *(const float4*)(mrow + lane * 4);
    atomicAdd(&orow[lane * 4 + 0], mv.x * sc);
    atomicAdd(&orow[lane * 4 + 1], mv.y * sc);
    atomicAdd(&orow[lane * 4 + 2], mv.z * sc);
    atomicAdd(&orow[lane * 4 + 3], mv.w * sc);
}

extern "C" void kernel_launch(void* const* d_in, const int* in_sizes, int n_in,
                              void* d_out, int out_size, void* d_ws, size_t ws_size,
                              hipStream_t stream)
{
    (void)n_in; (void)out_size; (void)ws_size;

    const float* des   = (const float*)d_in[0];
    const float* tweet = (const float*)d_in[1];
    const float* num   = (const float*)d_in[2];
    const float* cat   = (const float*)d_in[3];
    const float* nf    = (const float*)d_in[4];
    const int*   eidx  = (const int*)d_in[5];
    const int*   etype = (const int*)d_in[6];
    const float* W_des = (const float*)d_in[7];  const float* b_des = (const float*)d_in[8];
    const float* W_tw  = (const float*)d_in[9];  const float* b_tw  = (const float*)d_in[10];
    const float* W_num = (const float*)d_in[11]; const float* b_num = (const float*)d_in[12];
    const float* W_cat = (const float*)d_in[13]; const float* b_cat = (const float*)d_in[14];
    const float* W_new = (const float*)d_in[15]; const float* b_new = (const float*)d_in[16];
    const float* W_in  = (const float*)d_in[17]; const float* b_in  = (const float*)d_in[18];
    const float* W_root= (const float*)d_in[19];
    const float* W_rel = (const float*)d_in[20];
    const float* b_rg  = (const float*)d_in[21];
    const float* W_o1  = (const float*)d_in[22]; const float* b_o1  = (const float*)d_in[23];
    const float* W_o2  = (const float*)d_in[24]; const float* b_o2  = (const float*)d_in[25];

    const int Nn = in_sizes[2] / 7;   // nodes
    const int Ee = in_sizes[6];       // edges
    const int* src = eidx;
    const int* dst = eidx + Ee;

    // ---- workspace layout ----
    float* bufA = (float*)d_ws;                   // N*128 f32
    float* bufB = bufA + (size_t)Nn * 128;
    float* xr0  = bufB + (size_t)Nn * 128;
    float* xr1  = xr0  + (size_t)Nn * 128;
    float* inv  = xr1  + (size_t)Nn * 128;        // 2*N f32
    __bf16* wbase  = (__bf16*)(inv + 2 * (size_t)Nn);
    __bf16* WtDes  = wbase;                       // 32  x 768
    __bf16* WtTw   = WtDes  + (size_t)32  * 768;  // 48  x 768
    __bf16* WtIn   = WtTw   + (size_t)48  * 768;  // 128 x 128
    __bf16* WtRoot = WtIn   + (size_t)128 * 128;  // 4 x 128 x 128
    __bf16* WtRel  = WtRoot + (size_t)4 * 128 * 128;  // 8 x 128 x 128
    __bf16* WtO1   = WtRel  + (size_t)8 * 128 * 128;  // 128 x 128
    __bf16* WtO2   = WtO1   + (size_t)128 * 128;      // 16 x 128

    const dim3 blk(256);
    const int wpb = 256 / 32;
    const int rowTiles = (Nn + 15) / 16;
    const int gx = (rowTiles + wpb - 1) / wpb;
    auto pgrid = [](int elems) { return dim3((elems + 255) / 256); };
    auto ldsB  = [](int NT, int K) { return (size_t)NT * 16 * (K + 8) * sizeof(__bf16); };

    // ---- weight prep (bf16, transposed, padded) ----
    prep_weight<<<pgrid(32 * 768), blk, 0, stream>>>(W_des, 768, 28, 32, WtDes);
    prep_weight<<<pgrid(48 * 768), blk, 0, stream>>>(W_tw,  768, 36, 48, WtTw);
    prep_weight<<<pgrid(128 * 128), blk, 0, stream>>>(W_in, 128, 128, 128, WtIn);
    for (int i = 0; i < 4; ++i)
        prep_weight<<<pgrid(128 * 128), blk, 0, stream>>>(W_root + (size_t)i * 128 * 128,
                                                          128, 128, 128, WtRoot + (size_t)i * 128 * 128);
    for (int i = 0; i < 8; ++i)
        prep_weight<<<pgrid(128 * 128), blk, 0, stream>>>(W_rel + (size_t)i * 128 * 128,
                                                          128, 128, 128, WtRel + (size_t)i * 128 * 128);
    prep_weight<<<pgrid(128 * 128), blk, 0, stream>>>(W_o1, 128, 128, 128, WtO1);
    prep_weight<<<pgrid(16 * 128), blk, 0, stream>>>(W_o2, 128, 2, 16, WtO2);

    // ---- degrees -> inverse counts ----
    zero_f32<<<pgrid(2 * Nn), blk, 0, stream>>>(inv, 2 * Nn);
    degree_kernel<<<pgrid(Ee), blk, 0, stream>>>(dst, etype, inv, Ee, Nn);
    inv_kernel<<<pgrid(2 * Nn), blk, 0, stream>>>(inv, 2 * Nn);

    // ---- encoder ----
    gemm_wmma<2, true><<<gx, blk, ldsB(2, 768), stream>>>(des, 768, Nn, 768, WtDes, 28,
                                                          nullptr, bufA, 128, 0, FUSE_NONE);
    gemm_wmma<3, false><<<gx, blk, 0, stream>>>(tweet, 768, Nn, 768, WtTw, 36,
                                                nullptr, bufA, 128, 28, FUSE_NONE);
    encode_small<<<pgrid(Nn), blk, 0, stream>>>(num, cat, nf, W_num, b_num, W_cat, b_cat,
                                                W_new, b_new, b_des, b_tw, bufA, Nn);
    gemm_wmma<8, true><<<gx, blk, ldsB(8, 128), stream>>>(bufA, 128, Nn, 128, WtIn, 128,
                                                          b_in, bufB, 128, 0, FUSE_BIAS_LRELU);

    // ---- 4 RGCN layers (ping-pong) ----
    float* cur = bufB;
    float* nxt = bufA;
    for (int i = 0; i < 4; ++i) {
        gemm_wmma<8, true><<<gx, blk, ldsB(8, 128), stream>>>(cur, 128, Nn, 128,
            WtRoot + (size_t)i * 128 * 128, 128, b_rg + i * 128, nxt, 128, 0, FUSE_BIAS);
        gemm_wmma<8, true><<<gx, blk, ldsB(8, 128), stream>>>(cur, 128, Nn, 128,
            WtRel + ((size_t)i * 2 + 0) * 128 * 128, 128, nullptr, xr0, 128, 0, FUSE_NONE);
        gemm_wmma<8, true><<<gx, blk, ldsB(8, 128), stream>>>(cur, 128, Nn, 128,
            WtRel + ((size_t)i * 2 + 1) * 128 * 128, 128, nullptr, xr1, 128, 0, FUSE_NONE);
        scatter_kernel<<<dim3((Ee + wpb - 1) / wpb), blk, 0, stream>>>(src, dst, etype,
                                                                      xr0, xr1, inv, nxt, Ee, Nn);
        float* t = cur; cur = nxt; nxt = t;
    }

    // ---- head ----
    gemm_wmma<8, true><<<gx, blk, ldsB(8, 128), stream>>>(cur, 128, Nn, 128, WtO1, 128,
                                                          b_o1, xr0, 128, 0, FUSE_BIAS_LRELU);
    gemm_wmma<1, true><<<gx, blk, ldsB(1, 128), stream>>>(xr0, 128, Nn, 128, WtO2, 2,
                                                          b_o2, (float*)d_out, 2, 0, FUSE_BIAS);
}
